// SwinEncoder_31756988187271
// MI455X (gfx1250) — compile-verified, matrix operand order checked by reading, not run on hardware
//
#include <hip/hip_runtime.h>
#include <stdint.h>

typedef _Float16 h16;
typedef __attribute__((ext_vector_type(16))) _Float16 v16h;
typedef __attribute__((ext_vector_type(8)))  float    v8f;

union FragA { v16h v; uint32_t u[8]; };
union FragC { v8f  v; float    f[8]; };

#if defined(__HIP_DEVICE_COMPILE__) && \
    __has_builtin(__builtin_amdgcn_global_load_async_to_lds_b128) && \
    __has_builtin(__builtin_amdgcn_s_wait_asynccnt)
#define HAVE_ASYNC_LDS 1
typedef int v4i_t __attribute__((vector_size(16)));
typedef __attribute__((address_space(1))) v4i_t g_v4i;   // global AS vec4i
typedef __attribute__((address_space(3))) v4i_t l_v4i;   // LDS AS vec4i
#else
#define HAVE_ASYNC_LDS 0
#endif

#define LDA_A 40   // A-tile LDS row stride in halfs (80B: 16B-aligned rows for b128)
#define LDA_B 34   // B-tile LDS row stride in halfs (bank-conflict-free frag reads)

// K index inside a 32-wide K-slice for operand VGPR v (0..7), half hsel (0/1):
// VGPR0..3 -> K = 2v + 8*hsel ; VGPR4..7 -> K = 2v + 8 + 8*hsel   (pairs contiguous)
__device__ __forceinline__ int kmap(int v, int hsel) {
    return 2 * v + (v >= 4 ? 8 : 0) + hsel * 8;
}

__device__ __forceinline__ v8f wmma_f16(v16h a, v16h b, v8f c) {
    return __builtin_amdgcn_wmma_f32_16x16x32_f16(false, a, false, b, (short)0, c, false, false);
}

// ---------------------------------------------------------------------------
// Patch embed: 4x4 stride-4 conv of 1-channel image -> X[(b,y,x), 96] f32
// ---------------------------------------------------------------------------
__global__ __launch_bounds__(256) void patch_embed_k(
    const float* __restrict__ x, const float* __restrict__ w,
    const float* __restrict__ bias, float* __restrict__ X, int total)
{
    int idx = blockIdx.x * 256 + threadIdx.x;
    if (idx >= total) return;
    int c = idx % 96;
    int t = idx / 96;
    int b  = t / (112 * 112);
    int rr = t % (112 * 112);
    int y = rr / 112, xx = rr % 112;
    const float* px = x + ((size_t)b * 448 + y * 4) * 448 + xx * 4;
    float s = bias[c];
#pragma unroll
    for (int i = 0; i < 4; i++)
#pragma unroll
        for (int j = 0; j < 4; j++)
            s += px[i * 448 + j] * w[c * 16 + i * 4 + j];
    X[(size_t)t * 96 + c] = s;
}

// ---------------------------------------------------------------------------
// window-row -> source-token mapping (shift folded in). nWw = 1<<lw windows
// per image row; all divides are by constants or shifts.
// ---------------------------------------------------------------------------
__device__ __forceinline__ size_t win_row_to_token(
    int row, int lw, int Hh, int Ww, int shift)
{
    int bb  = (row >> (2 * lw)) / 49;                // row / (49*nWw^2)
    int rr  = row - bb * (49 << (2 * lw));
    int win = rr / 49;
    int pos = rr - win * 49;
    int wi = win >> lw, wj = win & ((1 << lw) - 1);
    int y = wi * 7 + pos / 7 + shift; if (y >= Hh) y -= Hh;
    int x = wj * 7 + pos % 7 + shift; if (x >= Ww) x -= Ww;
    return ((size_t)bb * Hh + y) * Ww + x;
}

// ---------------------------------------------------------------------------
// LayerNorm (f32 -> f16). windowed=1: output rows are window-partitioned with
// cyclic shift folded into the gather. One wave per output row.
// ---------------------------------------------------------------------------
__global__ __launch_bounds__(256) void ln_f16_k(
    const float* __restrict__ X, const float* __restrict__ g,
    const float* __restrict__ b, h16* __restrict__ Out,
    int T, int C, int windowed, int Hh, int Ww, int shift, int lw)
{
    int r = blockIdx.x * 8 + (threadIdx.x >> 5);
    int lane = threadIdx.x & 31;
    if (r >= T) return;
    size_t src = windowed ? win_row_to_token(r, lw, Hh, Ww, shift) : (size_t)r;
    const float* xr = X + src * C;
    float sum = 0.f, sq = 0.f;
    for (int c = lane; c < C; c += 32) { float v = xr[c]; sum += v; sq += v * v; }
    for (int d = 1; d < 32; d <<= 1) {
        sum += __shfl_xor(sum, d, 32);
        sq  += __shfl_xor(sq,  d, 32);
    }
    float mu = sum / C;
    float var = sq / C - mu * mu;
    float rs = rsqrtf(var + 1e-5f);
    h16* orow = Out + (size_t)r * C;
    for (int c = lane; c < C; c += 32)
        orow[c] = (h16)((xr[c] - mu) * rs * g[c] + b[c]);
}

// ---------------------------------------------------------------------------
// Patch merging gather + LayerNorm over 4C -> f16. One wave per output row.
// Feature block q: 0->(0,0) 1->(1,0) 2->(0,1) 3->(1,1)
// ---------------------------------------------------------------------------
__global__ __launch_bounds__(256) void merge_ln_k(
    const float* __restrict__ X, const float* __restrict__ g,
    const float* __restrict__ b, h16* __restrict__ Out,
    int Hh, int Ww, int C, int T2)
{
    int r = blockIdx.x * 8 + (threadIdx.x >> 5);
    int lane = threadIdx.x & 31;
    if (r >= T2) return;
    int H2 = Hh / 2, W2 = Ww / 2;
    int bb = r / (H2 * W2);
    int rr = r - bb * H2 * W2;
    int y2 = rr / W2, x2 = rr % W2;
    int C4 = 4 * C;
    const float* base = X + ((size_t)bb * Hh) * Ww * C;
    float sum = 0.f, sq = 0.f;
    for (int f = lane; f < C4; f += 32) {
        int q = f / C, c = f - q * C;
        int dy = q & 1, dx = q >> 1;
        float v = base[(((size_t)(2 * y2 + dy)) * Ww + 2 * x2 + dx) * C + c];
        sum += v; sq += v * v;
    }
    for (int d = 1; d < 32; d <<= 1) {
        sum += __shfl_xor(sum, d, 32);
        sq  += __shfl_xor(sq,  d, 32);
    }
    float mu = sum / C4;
    float var = sq / C4 - mu * mu;
    float rs = rsqrtf(var + 1e-5f);
    h16* orow = Out + (size_t)r * C4;
    for (int f = lane; f < C4; f += 32) {
        int q = f / C, c = f - q * C;
        int dy = q & 1, dx = q >> 1;
        float v = base[(((size_t)(2 * y2 + dy)) * Ww + 2 * x2 + dx) * C + c];
        orow[f] = (h16)((v - mu) * rs * g[f] + b[f]);
    }
}

// ---------------------------------------------------------------------------
// Tiled WMMA GEMM: C[M,N] = A[M,K](f16,row) * B[K,N](f32 weights, row) (+epi)
// 256 threads = 8 waves; block tile 256x128; each wave owns 64(M)x64(N):
// 16 wmma per 8 fragment loads per K-step.
// A tile: async global->LDS b128 (double-buffered, ASYNCcnt pipelined) when
// the builtin exists; otherwise register-prefetch. B tile (f32 weights) is
// register-staged with f32->f16 conversion + global_prefetch of the next tile.
// ---------------------------------------------------------------------------
#define E_F16B 0
#define E_GELU 1
#define E_REST 2
#define E_RESW 3
#define E_F32S 4

template <int EPI>
__global__ __launch_bounds__(256) void gemm_wmma_k(
    const h16* __restrict__ A, const float* __restrict__ Bw,
    const float* __restrict__ bias, h16* __restrict__ OutH,
    float* __restrict__ Xres, int M, int N, int K,
    int Hh, int Ww, int shift, int lw)
{
    __shared__ h16 As[2][256 * LDA_A];
    __shared__ h16 Bs[128 * LDA_B];
    const int tid = threadIdx.x;
    const int lane = tid & 31, wv = tid >> 5;
    const int wm = wv & 3, wn = wv >> 2;
    const int bm0 = blockIdx.y * 256, bn0 = blockIdx.x * 128;
    const int hsel = lane >> 4, nn = lane & 15;

    FragC acc[4][4];
#pragma unroll
    for (int ti = 0; ti < 4; ti++)
#pragma unroll
        for (int tj = 0; tj < 4; tj++)
#pragma unroll
            for (int r = 0; r < 8; r++) acc[ti][tj].f[r] = 0.0f;

    const int gr0 = bm0 + tid;           // one full 32-half A row per thread
    const bool grok = (gr0 < M);
    const int gr = grok ? gr0 : (M - 1); // clamped (garbage rows masked in epi)
    const int bcol = tid & 127;
    const int kq = (tid >> 7) * 16;
    const int gc = bn0 + bcol;

    h16 bh[16];
    auto loadB = [&](int k0) {
        if (gc < N) {
            const float* bsrc = Bw + (size_t)(k0 + kq) * N + gc;
            __builtin_prefetch(bsrc + 32 * (size_t)N, 0, 1); // next K-tile
#pragma unroll
            for (int i = 0; i < 16; i++) bh[i] = (h16)bsrc[(size_t)i * N];
        } else {
#pragma unroll
            for (int i = 0; i < 16; i++) bh[i] = (h16)0.f;
        }
    };
    auto commitB = [&]() {
        uint32_t* dstB = (uint32_t*)&Bs[bcol * LDA_B + kq];
#pragma unroll
        for (int i = 0; i < 8; i++) dstB[i] = ((const uint32_t*)bh)[i];
    };
    auto compute = [&](const h16* Ab) {
        FragA af[4], bf[4];
#pragma unroll
        for (int ti = 0; ti < 4; ti++) {
            const h16* base = &Ab[(wm * 64 + ti * 16 + nn) * LDA_A];
#pragma unroll
            for (int v = 0; v < 8; v++)
                af[ti].u[v] = *(const uint32_t*)&base[kmap(v, hsel)];
        }
#pragma unroll
        for (int tj = 0; tj < 4; tj++) {
            const h16* base = &Bs[(wn * 64 + tj * 16 + nn) * LDA_B];
#pragma unroll
            for (int v = 0; v < 8; v++)
                bf[tj].u[v] = *(const uint32_t*)&base[kmap(v, hsel)];
        }
#pragma unroll
        for (int ti = 0; ti < 4; ti++)
#pragma unroll
            for (int tj = 0; tj < 4; tj++)
                acc[ti][tj].v = wmma_f16(af[ti].v, bf[tj].v, acc[ti][tj].v);
    };

    const int nK = K >> 5;

#if HAVE_ASYNC_LDS
    // Async global->LDS A-tile pipeline: 4 x b128 per thread per K-tile.
    auto issueA = [&](int k0, int buf) {
        const h16* gsrc = A + (size_t)gr * K + k0;
        h16* dl = &As[buf][tid * LDA_A];
#pragma unroll
        for (int i = 0; i < 4; i++)
            __builtin_amdgcn_global_load_async_to_lds_b128(
                (g_v4i*)(uintptr_t)(gsrc + 8 * i),
                (l_v4i*)(uint32_t)(uintptr_t)(dl + 8 * i), 0, 0);
    };
    issueA(0, 0);
    loadB(0);
    for (int kt = 0; kt < nK; kt++) {
        const int buf = kt & 1;
        commitB();
        if (kt + 1 < nK) {
            issueA((kt + 1) << 5, buf ^ 1);       // overlaps wmma below
            __builtin_amdgcn_s_wait_asynccnt(4);  // retire tile kt's 4 ops
        } else {
            __builtin_amdgcn_s_wait_asynccnt(0);
        }
        __syncthreads();
        if (kt + 1 < nK) loadB((kt + 1) << 5);
        compute(&As[buf][0]);
        __syncthreads();
    }
#else
    // Fallback: register-prefetch A staging.
    uint32_t aw[16];
    auto loadA = [&](int k0) {
        if (grok) {
            const uint32_t* src = (const uint32_t*)(A + (size_t)gr * K + k0);
#pragma unroll
            for (int i = 0; i < 16; i++) aw[i] = src[i];
        } else {
#pragma unroll
            for (int i = 0; i < 16; i++) aw[i] = 0u;
        }
    };
    loadA(0);
    loadB(0);
    for (int kt = 0; kt < nK; kt++) {
        const int buf = kt & 1;
        uint32_t* dstA = (uint32_t*)&As[buf][tid * LDA_A];
#pragma unroll
        for (int i = 0; i < 16; i++) dstA[i] = aw[i];
        commitB();
        __syncthreads();
        if (kt + 1 < nK) { loadA((kt + 1) << 5); loadB((kt + 1) << 5); }
        compute(&As[buf][0]);
        __syncthreads();
    }
#endif

    // epilogue: row mapping hoisted out of the column loop
#pragma unroll
    for (int ti = 0; ti < 4; ti++) {
#pragma unroll
        for (int r = 0; r < 8; r++) {
            int row = bm0 + wm * 64 + ti * 16 + r + 8 * hsel;
            if (row >= M) continue;
            size_t obase;
            if (EPI == E_RESW) {
                obase = win_row_to_token(row, lw, Hh, Ww, shift) * (size_t)N;
            } else {
                obase = (size_t)row * N;
            }
#pragma unroll
            for (int tj = 0; tj < 4; tj++) {
                int col = bn0 + wn * 64 + tj * 16 + nn;
                if (col >= N) continue;
                float v = acc[ti][tj].f[r];
                if (EPI == E_F16B) {
                    OutH[obase + col] = (h16)(v + bias[col]);
                } else if (EPI == E_GELU) {
                    float t = v + bias[col];
                    OutH[obase + col] =
                        (h16)(0.5f * t * (1.0f + erff(t * 0.70710678118654752f)));
                } else if (EPI == E_REST || EPI == E_RESW) {
                    Xres[obase + col] += v + bias[col];
                } else { // E_F32S
                    Xres[obase + col] = v;
                }
            }
        }
    }
}

// ---------------------------------------------------------------------------
// Fused window attention: one block (4 waves) per (window, head). d = 32.
// S = Q*K^T (4 wmma, K=32), +rel-pos bias, softmax, O = P*V (4 wmma).
// All operand fragments are loaded into registers before the wmma bursts so
// the LDS waits amortize and no hazard NOPs are needed between wmma ops.
// ---------------------------------------------------------------------------
__global__ __launch_bounds__(128) void win_attn_k(
    const h16* __restrict__ QKV, const float* __restrict__ rpb,
    h16* __restrict__ O, int C, int nh)
{
    __shared__ h16 Qs[64 * 34];
    __shared__ h16 Ks[64 * 34];
    __shared__ h16 Vt[32 * 66];   // Vt[c][token]
    __shared__ h16 Ps[64 * 66];
    __shared__ float rpbh[256];   // 169 real + zero padding (idx <= 207 for pads)

    const int win = blockIdx.x, head = blockIdx.y;
    const int tid = threadIdx.x, lane = tid & 31, wv = tid >> 5;
    const int hsel = lane >> 4, nn = lane & 15;
    const int C3 = 3 * C;

    for (int i = tid; i < 256; i += 128)
        rpbh[i] = (i < 169) ? rpb[i * nh + head] : 0.0f;

    const int row = tid >> 1;
    const int part = (tid & 1) * 16;
    {
        uint32_t q[8], k[8];
        h16 vv[16];
        if (row < 49) {
            const h16* base = QKV + (size_t)(win * 49 + row) * C3 + head * 32 + part;
            const uint32_t* qp = (const uint32_t*)base;
            const uint32_t* kp = (const uint32_t*)(base + C);
            const h16* vp = base + 2 * C;
#pragma unroll
            for (int i = 0; i < 8; i++) { q[i] = qp[i]; k[i] = kp[i]; }
#pragma unroll
            for (int i = 0; i < 16; i++) vv[i] = vp[i];
        } else {
#pragma unroll
            for (int i = 0; i < 8; i++) { q[i] = 0u; k[i] = 0u; }
#pragma unroll
            for (int i = 0; i < 16; i++) vv[i] = (h16)0.f;
        }
        uint32_t* qd = (uint32_t*)&Qs[row * 34 + part];
        uint32_t* kd = (uint32_t*)&Ks[row * 34 + part];
#pragma unroll
        for (int i = 0; i < 8; i++) { qd[i] = q[i]; kd[i] = k[i]; }
#pragma unroll
        for (int i = 0; i < 16; i++) Vt[(part + i) * 66 + row] = vv[i];
    }
    __syncthreads();

    const int m0 = wv * 16;
    FragA aq, bk[4];
    {
        const h16* b = &Qs[(m0 + nn) * 34];
#pragma unroll
        for (int v = 0; v < 8; v++) aq.u[v] = *(const uint32_t*)&b[kmap(v, hsel)];
    }
#pragma unroll
    for (int cj = 0; cj < 4; cj++) {
        const h16* b = &Ks[(cj * 16 + nn) * 34];
#pragma unroll
        for (int v = 0; v < 8; v++) bk[cj].u[v] = *(const uint32_t*)&b[kmap(v, hsel)];
    }
    FragC sa[4];
    {
        v8f z = {0.f, 0.f, 0.f, 0.f, 0.f, 0.f, 0.f, 0.f};
#pragma unroll
        for (int cj = 0; cj < 4; cj++) sa[cj].v = wmma_f16(aq.v, bk[cj].v, z);
    }

    // per-lane column constants hoisted out of the row loop
    int jd[4], jm[4];
    bool jok[4];
#pragma unroll
    for (int cj = 0; cj < 4; cj++) {
        int j = cj * 16 + nn;
        jd[cj] = j / 7;
        jm[cj] = j - jd[cj] * 7;
        jok[cj] = (j < 49);
    }

    const float scale = 0.17677669529663687f; // 1/sqrt(32)
#pragma unroll
    for (int r = 0; r < 8; r++) {
        int i = m0 + r + 8 * hsel;
        int iy = i / 7, ixm = i % 7;
        float s[4];
#pragma unroll
        for (int cj = 0; cj < 4; cj++) {
            int ry = iy - jd[cj] + 6;
            int rx = ixm - jm[cj] + 6;
            // unconditional table read: rows i>=49 discarded; cols j>=49 masked
            float t = sa[cj].f[r] * scale + rpbh[ry * 13 + rx];
            s[cj] = jok[cj] ? t : -1e30f;
        }
        float mx = fmaxf(fmaxf(s[0], s[1]), fmaxf(s[2], s[3]));
        for (int d = 1; d < 16; d <<= 1) mx = fmaxf(mx, __shfl_xor(mx, d, 32));
        float sum = 0.f;
#pragma unroll
        for (int cj = 0; cj < 4; cj++) { s[cj] = __expf(s[cj] - mx); sum += s[cj]; }
        for (int d = 1; d < 16; d <<= 1) sum += __shfl_xor(sum, d, 32);
        float inv = 1.0f / sum;
#pragma unroll
        for (int cj = 0; cj < 4; cj++)
            Ps[(m0 + r + 8 * hsel) * 66 + cj * 16 + nn] = (h16)(s[cj] * inv);
    }
    __syncthreads();

    // O = P * V : load all fragments, then one wmma burst
    FragA ap[2], bv[2][2];
#pragma unroll
    for (int kc = 0; kc < 2; kc++) {
        const h16* pb = &Ps[(m0 + nn) * 66 + kc * 32];
#pragma unroll
        for (int v = 0; v < 8; v++) ap[kc].u[v] = *(const uint32_t*)&pb[kmap(v, hsel)];
#pragma unroll
        for (int cj = 0; cj < 2; cj++) {
            const h16* vb = &Vt[(cj * 16 + nn) * 66 + kc * 32];
#pragma unroll
            for (int v = 0; v < 8; v++) bv[kc][cj].u[v] = *(const uint32_t*)&vb[kmap(v, hsel)];
        }
    }
    FragC oa[2];
#pragma unroll
    for (int cj = 0; cj < 2; cj++)
#pragma unroll
        for (int r = 0; r < 8; r++) oa[cj].f[r] = 0.0f;
#pragma unroll
    for (int kc = 0; kc < 2; kc++)
#pragma unroll
        for (int cj = 0; cj < 2; cj++)
            oa[cj].v = wmma_f16(ap[kc].v, bv[kc][cj].v, oa[cj].v);

#pragma unroll
    for (int cj = 0; cj < 2; cj++)
#pragma unroll
        for (int r = 0; r < 8; r++) {
            int i = m0 + r + 8 * hsel;
            if (i < 49) {
                size_t grow = (size_t)win * 49 + i;
                O[grow * C + head * 32 + cj * 16 + nn] = (h16)oa[cj].f[r];
            }
        }
}

// ---------------------------------------------------------------------------
// Host orchestration
// ---------------------------------------------------------------------------
extern "C" void kernel_launch(void* const* d_in, const int* in_sizes, int n_in,
                              void* d_out, int out_size, void* d_ws, size_t ws_size,
                              hipStream_t stream)
{
    (void)in_sizes; (void)n_in; (void)ws_size;
    static const int DEPTH[4] = {2, 2, 6, 2};
    static const int NH[4]    = {3, 6, 12, 24};

    struct BP {
        const float *fc1_b, *fc1_w, *fc2_b, *fc2_w, *n1_b, *n1_g,
                    *n2_b, *n2_g, *proj_b, *proj_w, *qkv_b, *qkv_w, *rpb;
    };
    struct MP { const float *n_b, *n_g, *red_w; };

    // jax tree-flatten order: params{patch_b, patch_w, stages[blocks(sorted keys), merge]}, x
    int ix = 0;
    const float* patch_b = (const float*)d_in[ix++];
    const float* patch_w = (const float*)d_in[ix++];
    BP blk[12]; MP mg[3];
    int bc = 0, mc = 0;
    for (int si = 0; si < 4; si++) {
        for (int dd = 0; dd < DEPTH[si]; dd++) {
            BP p;
            p.fc1_b  = (const float*)d_in[ix++]; p.fc1_w  = (const float*)d_in[ix++];
            p.fc2_b  = (const float*)d_in[ix++]; p.fc2_w  = (const float*)d_in[ix++];
            p.n1_b   = (const float*)d_in[ix++]; p.n1_g   = (const float*)d_in[ix++];
            p.n2_b   = (const float*)d_in[ix++]; p.n2_g   = (const float*)d_in[ix++];
            p.proj_b = (const float*)d_in[ix++]; p.proj_w = (const float*)d_in[ix++];
            p.qkv_b  = (const float*)d_in[ix++]; p.qkv_w  = (const float*)d_in[ix++];
            p.rpb    = (const float*)d_in[ix++];
            blk[bc++] = p;
        }
        if (si < 3) {
            mg[mc].n_b   = (const float*)d_in[ix++];
            mg[mc].n_g   = (const float*)d_in[ix++];
            mg[mc].red_w = (const float*)d_in[ix++];
            mc++;
        }
    }
    const float* xin = (const float*)d_in[ix++];

    // workspace layout (all offsets 256B aligned)
    char* ws = (char*)d_ws;
    float* X   = (float*)ws;                         //  38,535,168 B f32 residual
    h16*   HW  = (h16*)(ws + 38535168);              //  19,267,584 B f16 LN out
    h16*   QKV = (h16*)(ws + 57802752);              //  77,070,336 B (union w/ HID)
    h16*   HID = QKV;
    h16*   ATTO= (h16*)(ws + 134873088);             //  19,267,584 B f16 attn out

    // patch embed
    {
        int total = 8 * 112 * 112 * 96;
        patch_embed_k<<<(total + 255) / 256, 256, 0, stream>>>(xin, patch_w, patch_b, X, total);
    }

    int Hh = 112, Ww = 112, C = 96;
    int lw = 4;                       // log2(windows per image row): 16,8,4,2
    int bi = 0, mi = 0;
    for (int si = 0; si < 4; si++) {
        int nh = NH[si];
        for (int dd = 0; dd < DEPTH[si]; dd++) {
            const BP& p = blk[bi++];
            int shift = (dd & 1) ? 3 : 0;
            int T = 8 * Hh * Ww;
            dim3 lnG((T + 7) / 8);

            // LN1 + shift + window partition -> HW (f16)
            ln_f16_k<<<lnG, 256, 0, stream>>>(X, p.n1_g, p.n1_b, HW, T, C, 1, Hh, Ww, shift, lw);

            // QKV GEMM: [T,C] x [C,3C] -> QKV f16
            {
                int M = T, N = 3 * C, K = C;
                dim3 g((N + 127) / 128, (M + 255) / 256);
                gemm_wmma_k<E_F16B><<<g, 256, 0, stream>>>(HW, p.qkv_w, p.qkv_b, QKV, nullptr,
                                                           M, N, K, Hh, Ww, 0, lw);
            }
            // window attention
            {
                dim3 g(T / 49, nh);
                win_attn_k<<<g, 128, 0, stream>>>(QKV, p.rpb, ATTO, C, nh);
            }
            // proj GEMM + inverse window/shift scatter + residual add into X
            {
                int M = T, N = C, K = C;
                dim3 g((N + 127) / 128, (M + 255) / 256);
                gemm_wmma_k<E_RESW><<<g, 256, 0, stream>>>(ATTO, p.proj_w, p.proj_b, nullptr, X,
                                                           M, N, K, Hh, Ww, shift, lw);
            }
            // LN2 -> HW
            ln_f16_k<<<lnG, 256, 0, stream>>>(X, p.n2_g, p.n2_b, HW, T, C, 0, Hh, Ww, 0, lw);
            // fc1 + GELU -> HID
            {
                int M = T, N = 4 * C, K = C;
                dim3 g((N + 127) / 128, (M + 255) / 256);
                gemm_wmma_k<E_GELU><<<g, 256, 0, stream>>>(HW, p.fc1_w, p.fc1_b, HID, nullptr,
                                                           M, N, K, Hh, Ww, 0, lw);
            }
            // fc2 + residual add into X
            {
                int M = T, N = C, K = 4 * C;
                dim3 g((N + 127) / 128, (M + 255) / 256);
                gemm_wmma_k<E_REST><<<g, 256, 0, stream>>>(HID, p.fc2_w, p.fc2_b, nullptr, X,
                                                           M, N, K, Hh, Ww, 0, lw);
            }
        }
        if (si < 3) {
            const MP& m = mg[mi++];
            int T2 = 8 * (Hh / 2) * (Ww / 2);
            merge_ln_k<<<(T2 + 7) / 8, 256, 0, stream>>>(X, m.n_g, m.n_b, HW, Hh, Ww, C, T2);
            int M = T2, N = 2 * C, K = 4 * C;
            dim3 g((N + 127) / 128, (M + 255) / 256);
            gemm_wmma_k<E_F32S><<<g, 256, 0, stream>>>(HW, m.red_w, nullptr, nullptr, X,
                                                       M, N, K, Hh, Ww, 0, lw);
            Hh /= 2; Ww /= 2; C *= 2; lw -= 1;
        }
    }

    (void)hipMemcpyAsync(d_out, X, (size_t)out_size * sizeof(float),
                         hipMemcpyDeviceToDevice, stream);
}
